// DBMBlock_68247030333810
// MI455X (gfx1250) — matrix operand order repaired and checked by reading, not compiled, not run on hardware
//
#include <hip/hip_runtime.h>
#include <hip/hip_bf16.h>
#include <stdint.h>

typedef __bf16 bf16_t;
typedef __attribute__((ext_vector_type(16))) __bf16 bf16x16;
typedef __attribute__((ext_vector_type(8)))  __bf16 bf16x8;
typedef __attribute__((ext_vector_type(8)))  float  f32x8;

#define DMODEL 768
#define DINNER 1536
#define DSTATE 16
#define DTRANK 48
#define NBATCH 2
#define LSEQ   1024
#define NROWS  (NBATCH * LSEQ)   // 2048

// ---------- helpers ----------
__device__ __forceinline__ bf16_t f2bf(float f) {
    unsigned u = __builtin_bit_cast(unsigned, f);
    unsigned r = u + 0x7FFFu + ((u >> 16) & 1u);   // round-to-nearest-even
    unsigned short h = (unsigned short)(r >> 16);
    return __builtin_bit_cast(bf16_t, h);
}

__device__ __forceinline__ float silu(float v) {
    return v / (1.0f + __expf(-v));
}

// ---------- fp32 -> bf16 conversion ----------
__global__ void cvt_bf16_kernel(const float* __restrict__ in,
                                bf16_t* __restrict__ out, int n) {
    int tid = blockIdx.x * blockDim.x + threadIdx.x;
    if (tid < n) out[tid] = f2bf(in[tid]);
}

__global__ void fill_zero_bf16_kernel(bf16_t* __restrict__ out, int n) {
    int tid = blockIdx.x * blockDim.x + threadIdx.x;
    if (tid < n) out[tid] = __builtin_bit_cast(bf16_t, (unsigned short)0);
}

// copy rows with column padding to zero (used for dt_w K=48 -> 64 and dtx)
__global__ void pack_pad_bf16_kernel(const float* __restrict__ in,
                                     bf16_t* __restrict__ out,
                                     int rows, int in_ld, int cols_copy, int out_ld) {
    int tid = blockIdx.x * blockDim.x + threadIdx.x;
    if (tid >= rows * out_ld) return;
    int c = tid % out_ld, r = tid / out_ld;
    float v = (c < cols_copy) ? in[(size_t)r * in_ld + c] : 0.0f;
    out[tid] = f2bf(v);
}

// ---------- WMMA GEMM:  Out(MxN) = A(MxK,bf16) * W(NxK,bf16)^T (+bias)(+act) ----------
// Each wave: 16x64 tile, 4 straight-line accumulators, 2-stage software pipeline
// (K must be a multiple of 64; all call sites use K in {64, 768, 1536}).
// W buffer must have >= (ceil(N/64)*64) valid (readable) rows; stores are guarded.
template<int OUT_BF16, int ACT>
__global__ __launch_bounds__(128)
void gemm_wmma_kernel(const bf16_t* __restrict__ A, int lda,
                      const bf16_t* __restrict__ W, int ldw,
                      const float* __restrict__ bias,
                      void* __restrict__ OutP, int ldc,
                      int M, int N, int K, int accumulate) {
    const int lane = threadIdx.x & 31;
    const int wave = threadIdx.x >> 5;
    const int half = lane >> 4;         // 0: lanes 0-15, 1: lanes 16-31
    const int lm   = lane & 15;
    const int mtile  = blockIdx.y * 64 + wave * 16;
    const int nblock = blockIdx.x * 64;

    // A-fragment K runs: contiguous K = 16*half + s (matches W fragment)
    const int run0 = half * 16;         // lane<16: K0-7   ; lane>=16: K16-23
    const int run1 = half * 16 + 8;     // lane<16: K8-15  ; lane>=16: K24-31
    const bf16_t* Arow = A + (size_t)(mtile + lm) * lda;
    const bf16_t* W0 = W + (size_t)(nblock +  0 + lm) * ldw + half * 16;
    const bf16_t* W1 = W + (size_t)(nblock + 16 + lm) * ldw + half * 16;
    const bf16_t* W2 = W + (size_t)(nblock + 32 + lm) * ldw + half * 16;
    const bf16_t* W3 = W + (size_t)(nblock + 48 + lm) * ldw + half * 16;

    f32x8 c0 = {}, c1 = {}, c2 = {}, c3 = {};

    // ---- 2-stage software pipeline over K (chunks of 32, unrolled x2) ----
    // Stage "p" registers hold chunk kb, stage "q" registers hold chunk kb+32.
    bf16x8 pa0 = *(const bf16x8*)(Arow + run0);
    bf16x8 pa1 = *(const bf16x8*)(Arow + run1);
    bf16x16 pb0 = *(const bf16x16*)(W0);
    bf16x16 pb1 = *(const bf16x16*)(W1);
    bf16x16 pb2 = *(const bf16x16*)(W2);
    bf16x16 pb3 = *(const bf16x16*)(W3);

    for (int kb = 0; kb < K; kb += 64) {
        const int k1 = kb + 32;
        __builtin_prefetch(Arow + k1, 0, 1);
        bf16x8  qa0 = *(const bf16x8*)(Arow + k1 + run0);
        bf16x8  qa1 = *(const bf16x8*)(Arow + k1 + run1);
        bf16x16 qb0 = *(const bf16x16*)(W0 + k1);
        bf16x16 qb1 = *(const bf16x16*)(W1 + k1);
        bf16x16 qb2 = *(const bf16x16*)(W2 + k1);
        bf16x16 qb3 = *(const bf16x16*)(W3 + k1);

        bf16x16 pa = __builtin_shufflevector(pa0, pa1,
                         0,1,2,3,4,5,6,7,8,9,10,11,12,13,14,15);
        c0 = __builtin_amdgcn_wmma_f32_16x16x32_bf16(false, pa, false, pb0, (short)0, c0, false, false);
        c1 = __builtin_amdgcn_wmma_f32_16x16x32_bf16(false, pa, false, pb1, (short)0, c1, false, false);
        c2 = __builtin_amdgcn_wmma_f32_16x16x32_bf16(false, pa, false, pb2, (short)0, c2, false, false);
        c3 = __builtin_amdgcn_wmma_f32_16x16x32_bf16(false, pa, false, pb3, (short)0, c3, false, false);

        // loads for the next iteration's first chunk (clamped, branch-free tail)
        const int k2 = (kb + 64 < K) ? kb + 64 : 0;
        pa0 = *(const bf16x8*)(Arow + k2 + run0);
        pa1 = *(const bf16x8*)(Arow + k2 + run1);
        pb0 = *(const bf16x16*)(W0 + k2);
        pb1 = *(const bf16x16*)(W1 + k2);
        pb2 = *(const bf16x16*)(W2 + k2);
        pb3 = *(const bf16x16*)(W3 + k2);

        bf16x16 qa = __builtin_shufflevector(qa0, qa1,
                         0,1,2,3,4,5,6,7,8,9,10,11,12,13,14,15);
        c0 = __builtin_amdgcn_wmma_f32_16x16x32_bf16(false, qa, false, qb0, (short)0, c0, false, false);
        c1 = __builtin_amdgcn_wmma_f32_16x16x32_bf16(false, qa, false, qb1, (short)0, c1, false, false);
        c2 = __builtin_amdgcn_wmma_f32_16x16x32_bf16(false, qa, false, qb2, (short)0, c2, false, false);
        c3 = __builtin_amdgcn_wmma_f32_16x16x32_bf16(false, qa, false, qb3, (short)0, c3, false, false);
    }

    f32x8 cc[4] = {c0, c1, c2, c3};
    #pragma unroll
    for (int i = 0; i < 4; ++i) {
        int nb = nblock + 16 * i;
        if (nb >= N) continue;                 // store-side guard only
        int n = nb + lm;
        float bv = bias ? bias[n] : 0.0f;
        #pragma unroll
        for (int r = 0; r < 8; ++r) {
            int mm = mtile + half * 8 + r;     // D layout: VGPR r -> M = r + 8*half
            float v = cc[i][r] + bv;
            if (ACT == 1) {                    // softplus
                float e = __expf(v);
                v = (v > 20.0f) ? v : __logf(1.0f + e);
            }
            size_t idx = (size_t)mm * ldc + n;
            if (OUT_BF16) {
                ((bf16_t*)OutP)[idx] = f2bf(v);
            } else {
                float* O = (float*)OutP;
                O[idx] = accumulate ? (O[idx] + v) : v;
            }
        }
    }
}

// ---------- causal (or anti-causal) depthwise conv1d + SiLU ----------
// xz: (NROWS, 2*DINNER) fp32, xc = cols [0, DINNER)
__global__ void conv_silu_kernel(const float* __restrict__ xz,
                                 const float* __restrict__ w,     // (DINNER,4)
                                 const float* __restrict__ cb,    // (DINNER)
                                 float* __restrict__ u, bf16_t* __restrict__ u_bf,
                                 int backward) {
    int tid = blockIdx.x * blockDim.x + threadIdx.x;
    if (tid >= NROWS * DINNER) return;
    int d = tid % DINNER;
    int row = tid / DINNER;
    int t = row % LSEQ;
    int b = row / LSEQ;
    float acc = cb[d];
    #pragma unroll
    for (int k = 0; k < 4; ++k) {
        int tt = backward ? (t + 3 - k) : (t + k - 3);
        if (tt >= 0 && tt < LSEQ)
            acc += w[d * 4 + k] * xz[((size_t)(b * LSEQ + tt)) * (2 * DINNER) + d];
    }
    float s = silu(acc);
    u[tid] = s;
    u_bf[tid] = f2bf(s);
}

// ---------- selective-scan: 16 lanes per (b,d) channel, one state per lane ----------
__global__ void scan_kernel(const float* __restrict__ dt,     // (NROWS, DINNER) post-softplus
                            const float* __restrict__ u,      // (NROWS, DINNER)
                            const float* __restrict__ proj,   // (NROWS, 80): B@48, C@64
                            const float* __restrict__ xz,     // z = cols [DINNER,2*DINNER)
                            const float* __restrict__ A_log,  // (DINNER, DSTATE)
                            const float* __restrict__ Dp,     // (DINNER)
                            float* __restrict__ y, bf16_t* __restrict__ y_bf,
                            int backward) {
    int gtid = blockIdx.x * blockDim.x + threadIdx.x;
    int lane = gtid & 31;
    int waveId = gtid >> 5;
    int i  = lane & 15;                      // state index
    int ch = waveId * 2 + (lane >> 4);       // channel = (b, d)
    if (ch >= NBATCH * DINNER) return;
    int b = ch / DINNER, d = ch % DINNER;

    float Ai = -__expf(A_log[d * DSTATE + i]);
    float Dv = Dp[d];
    float h = 0.0f;

    for (int s = 0; s < LSEQ; ++s) {
        int t = backward ? (LSEQ - 1 - s) : s;
        size_t row = (size_t)b * LSEQ + t;
        size_t rd  = row * DINNER + d;
        float dtv = dt[rd];
        float uv  = u[rd];
        const float* pr = proj + row * 80;
        float Bi = pr[48 + i];
        float Ci = pr[64 + i];
        h = __expf(dtv * Ai) * h + dtv * Bi * uv;
        float part = h * Ci;
        part += __shfl_xor(part, 8, 16);
        part += __shfl_xor(part, 4, 16);
        part += __shfl_xor(part, 2, 16);
        part += __shfl_xor(part, 1, 16);
        if (i == 0) {
            float zv = xz[row * (2 * DINNER) + DINNER + d];
            float yv = (part + Dv * uv) * silu(zv);
            y[rd] = yv;
            y_bf[rd] = f2bf(yv);
        }
    }
}

// ---------- residual + LayerNorm ----------
__global__ void residual_ln_kernel(const float* __restrict__ tmp,
                                   const float* __restrict__ x,
                                   const float* __restrict__ g,
                                   const float* __restrict__ bb,
                                   float* __restrict__ out) {
    int row = blockIdx.x;
    const float* t  = tmp + (size_t)row * DMODEL;
    const float* xr = x   + (size_t)row * DMODEL;
    __shared__ float s1[256], s2[256];
    float lsum = 0.0f, lsq = 0.0f;
    for (int j = threadIdx.x; j < DMODEL; j += blockDim.x) {
        float h = t[j] + xr[j];
        lsum += h; lsq += h * h;
    }
    s1[threadIdx.x] = lsum; s2[threadIdx.x] = lsq;
    __syncthreads();
    for (int off = 128; off > 0; off >>= 1) {
        if (threadIdx.x < off) {
            s1[threadIdx.x] += s1[threadIdx.x + off];
            s2[threadIdx.x] += s2[threadIdx.x + off];
        }
        __syncthreads();
    }
    float mean = s1[0] / (float)DMODEL;
    float var  = s2[0] / (float)DMODEL - mean * mean;
    float inv  = rsqrtf(var + 1e-5f);
    for (int j = threadIdx.x; j < DMODEL; j += blockDim.x) {
        float h = t[j] + xr[j];
        out[(size_t)row * DMODEL + j] = (h - mean) * inv * g[j] + bb[j];
    }
}

// ---------- host ----------
extern "C" void kernel_launch(void* const* d_in, const int* in_sizes, int n_in,
                              void* d_out, int out_size, void* d_ws, size_t ws_size,
                              hipStream_t stream) {
    (void)in_sizes; (void)n_in; (void)out_size; (void)ws_size;
    const float* x        = (const float*)d_in[0];
    const float* bm_in_w  = (const float*)d_in[1];
    const float* bm_in_b  = (const float*)d_in[2];
    const float* bm_out_w = (const float*)d_in[3];
    const float* bm_out_b = (const float*)d_in[4];
    const float* m_in_w   = (const float*)d_in[5];
    const float* m_conv_w = (const float*)d_in[6];
    const float* m_conv_b = (const float*)d_in[7];
    const float* m_xproj_w= (const float*)d_in[8];
    const float* m_dt_w   = (const float*)d_in[9];
    const float* m_dt_b   = (const float*)d_in[10];
    const float* m_A_log  = (const float*)d_in[11];
    const float* m_D      = (const float*)d_in[12];
    const float* m_out_w  = (const float*)d_in[13];
    const float* ln_g     = (const float*)d_in[14];
    const float* ln_b     = (const float*)d_in[15];
    float* out = (float*)d_out;

    char* p = (char*)d_ws;
    auto alloc = [&](size_t bytes) -> void* {
        void* r = (void*)p;
        p += (bytes + 255) & ~(size_t)255;
        return r;
    };
    bf16_t* xbf      = (bf16_t*)alloc((size_t)NROWS * DMODEL * 2);
    bf16_t* w_bm_in  = (bf16_t*)alloc((size_t)2 * DMODEL * DMODEL * 2);
    bf16_t* w_m_in   = (bf16_t*)alloc((size_t)2 * DINNER * DMODEL * 2);
    bf16_t* w_xproj  = (bf16_t*)alloc((size_t)128 * DINNER * 2);   // 80 rows + 48 zero pad
    bf16_t* w_dt     = (bf16_t*)alloc((size_t)DINNER * 64 * 2);
    bf16_t* w_m_out  = (bf16_t*)alloc((size_t)DMODEL * DINNER * 2);
    bf16_t* w_bm_out = (bf16_t*)alloc((size_t)DMODEL * DMODEL * 2);
    bf16_t* xp_bf    = (bf16_t*)alloc((size_t)NROWS * 2 * DMODEL * 2);
    float*  acc      = (float*) alloc((size_t)NROWS * DMODEL * 4);
    bf16_t* mo_bf    = (bf16_t*)alloc((size_t)NROWS * DMODEL * 2);
    float*  tmp      = (float*) alloc((size_t)NROWS * DMODEL * 4);
    float*  xz       = (float*) alloc((size_t)NROWS * 2 * DINNER * 4);
    float*  u        = (float*) alloc((size_t)NROWS * DINNER * 4);
    bf16_t* u_bf     = (bf16_t*)alloc((size_t)NROWS * DINNER * 2);
    float*  proj     = (float*) alloc((size_t)NROWS * 80 * 4);
    bf16_t* dtx_bf   = (bf16_t*)alloc((size_t)NROWS * 64 * 2);
    float*  dtbuf    = (float*) alloc((size_t)NROWS * DINNER * 4);
    float*  ybuf     = (float*) alloc((size_t)NROWS * DINNER * 4);
    bf16_t* y_bf     = (bf16_t*)alloc((size_t)NROWS * DINNER * 2);

    auto cvtN = [&](const float* in, bf16_t* o, int n) {
        cvt_bf16_kernel<<<(n + 255) / 256, 256, 0, stream>>>(in, o, n);
    };
    auto gemm = [&](int outBf, int act, const bf16_t* A, int lda,
                    const bf16_t* W, int ldw, const float* bias,
                    void* O, int ldc, int M, int N, int K, int accum) {
        dim3 grid((N + 63) / 64, M / 64), block(128);
        if (outBf)
            gemm_wmma_kernel<1,0><<<grid, block, 0, stream>>>(A, lda, W, ldw, bias, O, ldc, M, N, K, 0);
        else if (act)
            gemm_wmma_kernel<0,1><<<grid, block, 0, stream>>>(A, lda, W, ldw, bias, O, ldc, M, N, K, accum);
        else
            gemm_wmma_kernel<0,0><<<grid, block, 0, stream>>>(A, lda, W, ldw, bias, O, ldc, M, N, K, accum);
    };

    // weight / input conversion
    cvtN(x, xbf, NROWS * DMODEL);
    cvtN(bm_in_w, w_bm_in, 2 * DMODEL * DMODEL);
    cvtN(m_in_w, w_m_in, 2 * DINNER * DMODEL);
    cvtN(m_xproj_w, w_xproj, 80 * DINNER);
    {   int n = 48 * DINNER;   // zero-pad rows 80..127 so tail subtiles load valid data
        fill_zero_bf16_kernel<<<(n + 255) / 256, 256, 0, stream>>>(
            w_xproj + (size_t)80 * DINNER, n); }
    cvtN(m_out_w, w_m_out, DMODEL * DINNER);
    cvtN(bm_out_w, w_bm_out, DMODEL * DMODEL);
    {   int n = DINNER * 64;
        pack_pad_bf16_kernel<<<(n + 255) / 256, 256, 0, stream>>>(
            m_dt_w, w_dt, DINNER, DTRANK, DTRANK, 64); }

    // xp = x @ bm_in_w^T + b   (bf16 out, ld 1536)
    gemm(1, 0, xbf, DMODEL, w_bm_in, DMODEL, bm_in_b, xp_bf, 2 * DMODEL,
         NROWS, 2 * DMODEL, DMODEL, 0);

    for (int dir = 0; dir < 2; ++dir) {
        const bf16_t* xd = xp_bf + dir * DMODEL;   // fwd: cols 0..767, bwd: 768..1535
        // xz = x_d @ m_in_w^T   (fp32, 2048 x 3072)
        gemm(0, 0, xd, 2 * DMODEL, w_m_in, DMODEL, nullptr, xz, 2 * DINNER,
             NROWS, 2 * DINNER, DMODEL, 0);
        // u = silu(conv1d(xc))  (direction-aware padding)
        {   int n = NROWS * DINNER;
            conv_silu_kernel<<<(n + 255) / 256, 256, 0, stream>>>(
                xz, m_conv_w, m_conv_b, u, u_bf, dir); }
        // proj = u @ xproj_w^T  (2048 x 80)
        gemm(0, 0, u_bf, DINNER, w_xproj, DINNER, nullptr, proj, 80,
             NROWS, 80, DINNER, 0);
        // dtx (padded K=64) -> dt = softplus(dtx @ dt_w^T + dt_b)
        {   int n = NROWS * 64;
            pack_pad_bf16_kernel<<<(n + 255) / 256, 256, 0, stream>>>(
                proj, dtx_bf, NROWS, 80, DTRANK, 64); }
        gemm(0, 1, dtx_bf, 64, w_dt, 64, m_dt_b, dtbuf, DINNER,
             NROWS, DINNER, 64, 0);
        // selective scan + skip + gate (direction-aware)
        scan_kernel<<<192, 256, 0, stream>>>(dtbuf, u, proj, xz, m_A_log, m_D,
                                             ybuf, y_bf, dir);
        // acc (+)= y @ m_out_w^T
        gemm(0, 0, y_bf, DINNER, w_m_out, DINNER, nullptr, acc, DMODEL,
             NROWS, DMODEL, DINNER, dir);
    }

    // tmp = (out_f + out_b) @ bm_out_w^T + b
    cvtN(acc, mo_bf, NROWS * DMODEL);
    gemm(0, 0, mo_bf, DMODEL, w_bm_out, DMODEL, bm_out_b, tmp, DMODEL,
         NROWS, DMODEL, DMODEL, 0);

    // residual + layernorm
    residual_ln_kernel<<<NROWS, 256, 0, stream>>>(tmp, x, ln_g, ln_b, out);
}
